// GroupEncoder_91070486545133
// MI455X (gfx1250) — matrix operand order repaired and checked by reading, not compiled, not run on hardware
//
#include <hip/hip_runtime.h>
#include <hip/hip_bf16.h>

typedef __attribute__((ext_vector_type(16))) _Float16 v16h;
typedef __attribute__((ext_vector_type(8)))  float    v8f;

#define XS  80         // half-stride of 16x64 activation tiles: 160B/row, 16B aligned
#define WTS 72         // half-stride of transposed 64x64 weights: 144B/row, 16B aligned
#define WG  256

__device__ __forceinline__ v8f wmma16x16x32(v16h a, v16h b, v8f c) {
    // D = A(16x32 f16) * B(32x16 f16) + C(16x16 f32)
    return __builtin_amdgcn_wmma_f32_16x16x32_f16(
        /*neg_a=*/false, a, /*neg_b=*/false, b,
        /*c_mod=*/(short)0, c, /*reuse_a=*/false, /*reuse_b=*/false);
}

// A-matrix 16x32 f16 lane layout (ISA 7.12.2): lanes 0-15/16-31 both hold rows 0..15;
// lane hi-half holds K+8. half j -> K = (j<8 ? j : j+8) + 8*hi + 32*kc.
// Both 8-half runs are contiguous -> two ds_load_b128.
__device__ __forceinline__ v16h load_a_lds(const _Float16* buf, int lane, int kc) {
    int row = lane & 15, hi = lane >> 4;
    const _Float16* p = buf + row * XS + 8 * hi + 32 * kc;
    union { v16h v; uint4 q[2]; } u;
    u.q[0] = *(const uint4*)(p);        // K = 8hi .. 8hi+7   (+32kc)
    u.q[1] = *(const uint4*)(p + 16);   // K = 16+8hi .. 23+8hi
    return u.v;
}

// B-matrix 32x16 f16 lane layout: lane&15 = column, hi half-wave holds K+8 (same K order as A).
// wt is transposed [n][k] in LDS, so each lane reads two contiguous 8-half runs -> 2x b128.
__device__ __forceinline__ v16h load_b_ldsT(const _Float16* wt, int lane, int kc, int n0) {
    int col = lane & 15, hi = lane >> 4;
    const _Float16* p = wt + (n0 + col) * WTS + 8 * hi + 32 * kc;
    union { v16h v; uint4 q[2]; } u;
    u.q[0] = *(const uint4*)(p);
    u.q[1] = *(const uint4*)(p + 16);
    return u.v;
}

__device__ __forceinline__ float siluf(float v) {
    // v * sigmoid(v); fast v_rcp_f32 instead of IEEE divide
    return v * __builtin_amdgcn_rcpf(1.f + __expf(-v));
}
__device__ __forceinline__ float softplusf(float v) {
    return v > 20.f ? v : log1pf(__expf(v));
}

// ---------------------------------------------------------------- kernel 1: zero ws
__global__ __launch_bounds__(WG) void zero_f32(float* p, int n) {
    int i = blockIdx.x * WG + threadIdx.x;
    if (i < n) p[i] = 0.f;
}

// ------------------------------------------------- kernel 2: phi MLP + scatter-add
__global__ __launch_bounds__(WG) void phi_scatter(
    const float* __restrict__ x, const int* __restrict__ labels,
    const float* __restrict__ W1, const float* __restrict__ b1,
    const float* __restrict__ W2, const float* __restrict__ b2,
    float* __restrict__ gsum, float* __restrict__ gcnt, int Bn)
{
    __shared__ __align__(16) _Float16 W1sT[64 * WTS];
    __shared__ __align__(16) _Float16 W2sT[64 * WTS];
    __shared__ float b1s[64], b2s[64];
    __shared__ __align__(16) _Float16 xbuf[8][16 * XS];
    __shared__ __align__(16) _Float16 zbuf[8][16 * XS];
    __shared__ int lbl[8][16];

    const int tid = threadIdx.x;
    // stage weights transposed: W[k][n] (row-major global) -> WsT[n][k]
    for (int i = tid; i < 64 * 64; i += WG) {
        int k = i >> 6, n = i & 63;
        W1sT[n * WTS + k] = (_Float16)W1[i];
        W2sT[n * WTS + k] = (_Float16)W2[i];
    }
    if (tid < 64) { b1s[tid] = b1[tid]; b2s[tid] = b2[tid]; }
    __syncthreads();

    const int wave = tid >> 5, lane = tid & 31;
    const int col = lane & 15, hi = lane >> 4;
    const long long rowBase = (long long)blockIdx.x * 128 + wave * 16;

    if (lane < 16) {
        long long rr = rowBase + lane;
        lbl[wave][lane] = (rr < Bn) ? labels[rr] : -1;
    }
    // stage 16x64 f32 tile -> f16 LDS: b128 global loads, b64 LDS stores
    for (int it = 0; it < 8; ++it) {
        int idx = lane + it * 32;              // 256 float4 chunks
        int r = idx >> 4, cc = (idx & 15) * 4;
        long long rr = rowBase + r;
        float4 v = make_float4(0.f, 0.f, 0.f, 0.f);
        if (rr < Bn) v = *(const float4*)(x + rr * 64 + cc);
        union { _Float16 h[4]; uint2 u; } pk;
        pk.h[0] = (_Float16)v.x; pk.h[1] = (_Float16)v.y;
        pk.h[2] = (_Float16)v.z; pk.h[3] = (_Float16)v.w;
        *(uint2*)(&xbuf[wave][r * XS + cc]) = pk.u;
    }
    __builtin_amdgcn_wave_barrier();   // per-wave LDS ops are in-order; stop reordering

    // ---- layer 1: z = silu(x @ W1 + b1)
    v16h a0 = load_a_lds(&xbuf[wave][0], lane, 0);
    v16h a1 = load_a_lds(&xbuf[wave][0], lane, 1);
#pragma unroll
    for (int t = 0; t < 4; ++t) {
        const int n0 = t * 16;
        v8f c = {};
        c = wmma16x16x32(a0, load_b_ldsT(W1sT, lane, 0, n0), c);
        c = wmma16x16x32(a1, load_b_ldsT(W1sT, lane, 1, n0), c);
        float bv = b1s[n0 + col];
#pragma unroll
        for (int r = 0; r < 8; ++r) {
            float v = siluf(c[r] + bv);
            zbuf[wave][(r + 8 * hi) * XS + n0 + col] = (_Float16)v;  // C/D layout -> row-major
        }
    }
    __builtin_amdgcn_wave_barrier();

    // ---- layer 2: z2 = silu(z @ W2 + b2), scatter into group sums
    a0 = load_a_lds(&zbuf[wave][0], lane, 0);
    a1 = load_a_lds(&zbuf[wave][0], lane, 1);
#pragma unroll
    for (int t = 0; t < 4; ++t) {
        const int n0 = t * 16;
        v8f c = {};
        c = wmma16x16x32(a0, load_b_ldsT(W2sT, lane, 0, n0), c);
        c = wmma16x16x32(a1, load_b_ldsT(W2sT, lane, 1, n0), c);
        float bv = b2s[n0 + col];
#pragma unroll
        for (int r = 0; r < 8; ++r) {
            int m = r + 8 * hi;
            int g = lbl[wave][m];
            float v = siluf(c[r] + bv);
            if (g >= 0) atomicAdd(&gsum[(long long)g * 64 + n0 + col], v);
        }
    }
    if (lane < 16) {
        int g = lbl[wave][lane];
        if (g >= 0) atomicAdd(&gcnt[g], 1.f);
    }
}

// --------------------------------------- kernel 3: rho(group mean) + alpha/beta heads
__global__ __launch_bounds__(WG) void rho_heads(
    const float* __restrict__ gsum, const float* __restrict__ gcnt,
    const float* __restrict__ W3, const float* __restrict__ b3,
    const float* __restrict__ Wa, const float* __restrict__ ba,
    const float* __restrict__ Wb, const float* __restrict__ bb,
    float* __restrict__ out, int Kn)
{
    __shared__ __align__(16) _Float16 W3sT[64 * WTS];
    __shared__ float b3s[64], Was[64], Wbs[64];

    const int tid = threadIdx.x;
    for (int i = tid; i < 64 * 64; i += WG) {
        int k = i >> 6, n = i & 63;
        W3sT[n * WTS + k] = (_Float16)W3[i];
    }
    if (tid < 64) { b3s[tid] = b3[tid]; Was[tid] = Wa[tid]; Wbs[tid] = Wb[tid]; }
    __syncthreads();

    const int wave = tid >> 5, lane = tid & 31;
    const int g0 = (blockIdx.x * 8 + wave) * 16;
    if (g0 >= Kn) return;

    const int col = lane & 15, hi = lane >> 4;
    // load group-mean tile directly in A layout; all runs are b128s
    int grow = g0 + (lane & 15); if (grow > Kn - 1) grow = Kn - 1;
    float inv = __builtin_amdgcn_rcpf(fmaxf(gcnt[grow], 1.f));
    const float* gb = gsum + (long long)grow * 64 + 8 * hi;
    float4 q[8];
#pragma unroll
    for (int c2 = 0; c2 < 2; ++c2) {          // kc chunks (+0, +32)
        q[c2 * 4 + 0] = *(const float4*)(gb + 32 * c2);
        q[c2 * 4 + 1] = *(const float4*)(gb + 32 * c2 + 4);
        q[c2 * 4 + 2] = *(const float4*)(gb + 32 * c2 + 16);
        q[c2 * 4 + 3] = *(const float4*)(gb + 32 * c2 + 20);
    }
    v16h a0, a1;
#pragma unroll
    for (int j = 0; j < 4; ++j) {
        const float* f0 = (const float*)&q[0];
        const float* f1 = (const float*)&q[4];
        a0[j]      = (_Float16)(f0[j]      * inv);
        a0[j + 4]  = (_Float16)(f0[j + 4]  * inv);
        a0[j + 8]  = (_Float16)(f0[j + 8]  * inv);
        a0[j + 12] = (_Float16)(f0[j + 12] * inv);
        a1[j]      = (_Float16)(f1[j]      * inv);
        a1[j + 4]  = (_Float16)(f1[j + 4]  * inv);
        a1[j + 8]  = (_Float16)(f1[j + 8]  * inv);
        a1[j + 12] = (_Float16)(f1[j + 12] * inv);
    }

    float accA[8], accB[8];
#pragma unroll
    for (int r = 0; r < 8; ++r) { accA[r] = 0.f; accB[r] = 0.f; }

#pragma unroll
    for (int t = 0; t < 4; ++t) {
        const int n0 = t * 16;
        v8f c = {};
        c = wmma16x16x32(a0, load_b_ldsT(W3sT, lane, 0, n0), c);
        c = wmma16x16x32(a1, load_b_ldsT(W3sT, lane, 1, n0), c);
        float b3v = b3s[n0 + col], wav = Was[n0 + col], wbv = Wbs[n0 + col];
#pragma unroll
        for (int r = 0; r < 8; ++r) {
            float h = siluf(c[r] + b3v);
            accA[r] += h * wav;
            accB[r] += h * wbv;
        }
    }
    // reduce across the 16 column lanes (xor masks < 16 stay within half-waves)
#pragma unroll
    for (int off = 1; off < 16; off <<= 1) {
#pragma unroll
        for (int r = 0; r < 8; ++r) {
            accA[r] += __shfl_xor(accA[r], off, 32);
            accB[r] += __shfl_xor(accB[r], off, 32);
        }
    }
    if (col == 0) {
        float baV = ba[0], bbV = bb[0];
#pragma unroll
        for (int r = 0; r < 8; ++r) {
            int g = g0 + r + 8 * hi;
            if (g < Kn) {
                out[g]      = softplusf(accA[r] + baV) + 1e-4f;
                out[Kn + g] = softplusf(accB[r] + bbV) + 1e-4f;
            }
        }
    }
}

// ------------------------------------------------ kernel 4: Gamma(alpha)/beta sample
__device__ __forceinline__ float urand(unsigned& s) {
    s = s * 747796405u + 2891336453u;
    unsigned w = ((s >> ((s >> 28) + 4u)) ^ s) * 277803737u;
    w = (w >> 22) ^ w;
    return (float)(w >> 8) * (1.f / 16777216.f) + 1e-7f;
}

__global__ __launch_bounds__(WG) void gamma_sample(
    const float* __restrict__ out, float* __restrict__ tau, int Kn)
{
    int g = blockIdx.x * WG + threadIdx.x;
    if (g >= Kn) return;
    float alpha = out[g], beta = out[Kn + g];
    unsigned rng = ((unsigned)g * 2654435761u) ^ 0x9E3779B9u;
    float a = alpha, boost = 1.f;
    if (a < 1.f) { boost = powf(urand(rng), 1.f / a); a += 1.f; }
    float d = a - 1.f / 3.f;
    float c = rsqrtf(9.f * d);
    float sample = d;                       // fallback == mode-ish
    for (int it = 0; it < 24; ++it) {       // Marsaglia-Tsang, bounded & deterministic
        float u1 = urand(rng), u2 = urand(rng);
        float xn = sqrtf(-2.f * logf(u1)) * cosf(6.28318530718f * u2);
        float v = 1.f + c * xn;
        if (v <= 0.f) continue;
        v = v * v * v;
        float u = urand(rng);
        if (logf(u) < 0.5f * xn * xn + d - d * v + d * logf(v)) { sample = d * v; break; }
    }
    tau[g] = boost * sample / beta;
}

// ------------------------------------------------------- kernel 5: gather per refl
__global__ __launch_bounds__(WG) void gather_tau(
    const float* __restrict__ tau, const int* __restrict__ labels,
    float* __restrict__ outTau, int Bn, int Kn)
{
    int i = blockIdx.x * WG + threadIdx.x;
    if (i >= Bn) return;
    int g = labels[i];
    outTau[i] = (g >= 0 && g < Kn) ? tau[g] : 0.f;
}

extern "C" void kernel_launch(void* const* d_in, const int* in_sizes, int n_in,
                              void* d_out, int out_size, void* d_ws, size_t ws_size,
                              hipStream_t stream) {
    const float* x      = (const float*)d_in[0];
    const int*   labels = (const int*)  d_in[1];
    const float* W1 = (const float*)d_in[2];
    const float* b1 = (const float*)d_in[3];
    const float* W2 = (const float*)d_in[4];
    const float* b2 = (const float*)d_in[5];
    const float* W3 = (const float*)d_in[6];
    const float* b3 = (const float*)d_in[7];
    const float* Wa = (const float*)d_in[8];
    const float* ba = (const float*)d_in[9];
    const float* Wb = (const float*)d_in[10];
    const float* bb = (const float*)d_in[11];

    const int Bn = in_sizes[1];               // number of reflections
    const int Kn = (out_size - Bn) / 2;       // number of groups

    float* out  = (float*)d_out;              // [alpha(K) | beta(K) | tau(B)]
    float* gsum = (float*)d_ws;               // K*64
    float* gcnt = gsum + (size_t)Kn * 64;     // K
    float* tau  = gcnt + Kn;                  // K

    const int zn = Kn * 65;
    zero_f32<<<(zn + WG - 1) / WG, WG, 0, stream>>>(gsum, zn);

    phi_scatter<<<(Bn + 127) / 128, WG, 0, stream>>>(
        x, labels, W1, b1, W2, b2, gsum, gcnt, Bn);

    rho_heads<<<(Kn + 127) / 128, WG, 0, stream>>>(
        gsum, gcnt, W3, b3, Wa, ba, Wb, bb, out, Kn);

    gamma_sample<<<(Kn + WG - 1) / WG, WG, 0, stream>>>(out, tau, Kn);

    gather_tau<<<(Bn + WG - 1) / WG, WG, 0, stream>>>(tau, labels, out + 2 * (size_t)Kn, Bn, Kn);
}
